// LSTMCRFSeqLabeling_6305011990730
// MI455X (gfx1250) — matrix operand order repaired and checked by reading, not compiled, not run on hardware
//
#include <hip/hip_runtime.h>
#include <hip/hip_bf16.h>
#include <math.h>

// Problem constants (from reference)
#define HD 512      // hidden
#define BB 32       // batch
#define SS 256      // seq len
#define LL 20       // labels
#define G4 2048     // 4*H gate width

typedef __attribute__((ext_vector_type(16))) _Float16 v16h;
typedef __attribute__((ext_vector_type(8)))  _Float16 v8h;
typedef __attribute__((ext_vector_type(8)))  float    v8f;

// ---------------------------------------------------------------------------
// WMMA fragment helpers; A/B use this file's contiguous k order, C/D layout per ISA docs
// (CDNA5 ISA §7.12.2), wave32.
// ---------------------------------------------------------------------------

__device__ inline v8f vzero8f() {
  v8f v;
#pragma unroll
  for (int i = 0; i < 8; ++i) v[i] = 0.f;
  return v;
}

// A operand: 16x32 f16 tile of row-major A [.., ldk].
// lane 0-15: M=lane,   K = k0..k0+15
// lane 16-31: M=lane&15, K = k0+16..k0+31
__device__ inline v16h load_afrag(const _Float16* A, int ldk, int m0, int k0, int lane) {
  int m  = m0 + (lane & 15);
  int kb = k0 + ((lane >> 4) << 4);              // +0 or +16
  return *reinterpret_cast<const v16h*>(A + (size_t)m * ldk + kb);
}

// B operand: 32x16 f16 tile of B where B[k][n] = W[n][k], W row-major [N, ldk].
// lane 0-15: N=lane,   K = k0..k0+15  (16 contiguous f16 of row n -> one 32B load)
// lane 16-31: N=lane&15, K = k0+16..k0+31
__device__ inline v16h load_bfrag(const _Float16* W, int ldk, int n0, int k0, int lane) {
  int n  = n0 + (lane & 15);
  int kk = k0 + ((lane >> 4) << 4);              // +0 or +16
  return *reinterpret_cast<const v16h*>(W + (size_t)n * ldk + kk);
}

// C/D 16x16 f32: VGPR r -> M = r + (lane>=16 ? 8 : 0), N = lane&15
__device__ inline v8f load_cfrag(const float* C, int ldn, int m0, int n0, int lane) {
  int n  = n0 + (lane & 15);
  int mb = m0 + ((lane >> 4) << 3);
  v8f c;
#pragma unroll
  for (int r = 0; r < 8; ++r) c[r] = C[(size_t)(mb + r) * ldn + n];
  return c;
}

__device__ inline void store_cfrag(float* C, int ldn, int m0, int n0, int lane, v8f c) {
  int n  = n0 + (lane & 15);
  int mb = m0 + ((lane >> 4) << 3);
#pragma unroll
  for (int r = 0; r < 8; ++r) C[(size_t)(mb + r) * ldn + n] = c[r];
}

__device__ inline void store_cfrag_bias(float* C, int ldn, int m0, int n0, int lane,
                                        v8f c, const float* bias) {
  int n  = n0 + (lane & 15);
  int mb = m0 + ((lane >> 4) << 3);
  float bv = bias[n];
#pragma unroll
  for (int r = 0; r < 8; ++r) C[(size_t)(mb + r) * ldn + n] = c[r] + bv;
}

// ---------------------------------------------------------------------------
// Conversion kernels
// ---------------------------------------------------------------------------

// inputs [B,S,H] f32 -> x16 [S,B,H] f16 (time-major for the recurrence)
__global__ void k_x_transpose_f16(const float* __restrict__ in, _Float16* __restrict__ out) {
  int idx = blockIdx.x * blockDim.x + threadIdx.x;
  if (idx >= SS * BB * HD) return;
  int h = idx % HD;
  int sb = idx / HD;
  int b = sb % BB;
  int s = sb / BB;
  out[idx] = (_Float16)in[((size_t)b * SS + s) * HD + h];
}

__global__ void k_cvt_f16(const float* __restrict__ in, _Float16* __restrict__ out, int n) {
  int idx = blockIdx.x * blockDim.x + threadIdx.x;
  if (idx < n) out[idx] = (_Float16)in[idx];
}

__global__ void k_bias_sum(const float* __restrict__ a, const float* __restrict__ b,
                           float* __restrict__ out, int n) {
  int idx = blockIdx.x * blockDim.x + threadIdx.x;
  if (idx < n) out[idx] = a[idx] + b[idx];
}

// ---------------------------------------------------------------------------
// Batched input projection: out[M=8192, 2048] = A[M,K] @ W[2048,K]^T + bias
// One wave computes one 16-row M tile x 4 consecutive 16-col N tiles,
// reusing the A fragment across the 4 WMMAs per K chunk.
// ---------------------------------------------------------------------------
__global__ __launch_bounds__(256)
void k_gemm_xw(const _Float16* __restrict__ A, int K,
               const _Float16* __restrict__ W,
               const float* __restrict__ bias,
               float* __restrict__ out) {
  int lane = threadIdx.x & 31;
  int gw = blockIdx.x * (blockDim.x >> 5) + (threadIdx.x >> 5);
  int mtile = gw >> 5;                 // 0..511  (8192/16)
  int ngrp  = gw & 31;                 // 0..31   (128 ntiles / 4)
  int m0 = mtile * 16;
  int n0 = ngrp * 64;

  v8f acc0 = vzero8f(), acc1 = vzero8f(), acc2 = vzero8f(), acc3 = vzero8f();
  for (int k0 = 0; k0 < K; k0 += 32) {
    v16h a  = load_afrag(A, K, m0, k0, lane);
    v16h b0 = load_bfrag(W, K, n0,      k0, lane);
    v16h b1 = load_bfrag(W, K, n0 + 16, k0, lane);
    v16h b2 = load_bfrag(W, K, n0 + 32, k0, lane);
    v16h b3 = load_bfrag(W, K, n0 + 48, k0, lane);
    acc0 = __builtin_amdgcn_wmma_f32_16x16x32_f16(false, a, false, b0, (short)0, acc0, false, false);
    acc1 = __builtin_amdgcn_wmma_f32_16x16x32_f16(false, a, false, b1, (short)0, acc1, false, false);
    acc2 = __builtin_amdgcn_wmma_f32_16x16x32_f16(false, a, false, b2, (short)0, acc2, false, false);
    acc3 = __builtin_amdgcn_wmma_f32_16x16x32_f16(false, a, false, b3, (short)0, acc3, false, false);
  }
  store_cfrag_bias(out, G4, m0, n0,      lane, acc0, bias);
  store_cfrag_bias(out, G4, m0, n0 + 16, lane, acc1, bias);
  store_cfrag_bias(out, G4, m0, n0 + 32, lane, acc2, bias);
  store_cfrag_bias(out, G4, m0, n0 + 48, lane, acc3, bias);
}

// ---------------------------------------------------------------------------
// Persistent LSTM recurrence. grid = 2 blocks (fwd / bwd direction), 1024
// threads = 32 waves on one WGP. LDS holds g[32][2048] fp32 (256KB) + the
// f16 hidden state h[32][512] (32KB) = 288KB <= 320KB/WGP. c stays in regs.
// Per step: g = xw_t + h @ Whh^T via 256 WMMA tiles (8 per wave), then
// element-wise gates, new h written to LDS (next step's A operand) + global.
// ---------------------------------------------------------------------------
__global__ __launch_bounds__(1024)
void k_lstm_recur(const float* __restrict__ xw_f, const float* __restrict__ xw_b,
                  const _Float16* __restrict__ whh_f, const _Float16* __restrict__ whh_b,
                  _Float16* __restrict__ hout /* [S,B,2H] */) {
  extern __shared__ char smem[];
  float*    gS = (float*)smem;                                       // 32 x 2048 fp32
  _Float16* hS = (_Float16*)(smem + (size_t)BB * G4 * sizeof(float)); // 32 x 512 f16

  const int dir = blockIdx.x;                 // 0 = forward, 1 = backward
  const float* xw = dir ? xw_b : xw_f;
  const _Float16* W = dir ? whh_b : whh_f;
  const int hoff = dir ? HD : 0;

  int tid = threadIdx.x;
  int lane = tid & 31;
  int wid = tid >> 5;
  const int m  = tid >> 5;                    // gate-phase row (0..31)
  const int jl = tid & 31;                    // gate-phase col base

  for (int i = tid; i < BB * HD; i += 1024) hS[i] = (_Float16)0.f;
  float c[16];
#pragma unroll
  for (int q = 0; q < 16; ++q) c[q] = 0.f;
  __syncthreads();

  for (int stp = 0; stp < SS; ++stp) {
    int t = dir ? (SS - 1 - stp) : stp;
    const float* xwt = xw + (size_t)t * BB * G4;

    if (stp + 1 < SS) {                        // warm L2/L0 with next step's xw
      int tn = dir ? (t - 1) : (t + 1);
      __builtin_prefetch(xw + (size_t)tn * BB * G4 + (size_t)tid * 64, 0, 0);
    }

    // --- g = xw_t + h @ Whh^T : 8 output tiles per wave ---
    for (int tt = wid * 8; tt < wid * 8 + 8; ++tt) {
      int mt = (tt & 1) * 16;
      int nt = (tt >> 1) * 16;
      v8f acc = load_cfrag(xwt, G4, mt, nt, lane);   // C initialized from xw
      for (int k0 = 0; k0 < HD; k0 += 32) {
        v16h a = load_afrag(hS, HD, mt, k0, lane);   // LDS-resident h
        v16h b = load_bfrag(W, HD, nt, k0, lane);    // L2-resident Whh
        acc = __builtin_amdgcn_wmma_f32_16x16x32_f16(false, a, false, b, (short)0, acc, false, false);
      }
      store_cfrag(gS, G4, mt, nt, lane, acc);
    }
    __syncthreads();

    // --- element-wise gates (PyTorch order i,f,g,o) ---
#pragma unroll
    for (int q = 0; q < 16; ++q) {
      int j = jl + q * 32;
      float gi = gS[m * G4 + j];
      float gf = gS[m * G4 + HD + j];
      float gg = gS[m * G4 + 2 * HD + j];
      float go = gS[m * G4 + 3 * HD + j];
      float si = 1.f / (1.f + __expf(-gi));
      float sf = 1.f / (1.f + __expf(-gf));
      float so = 1.f / (1.f + __expf(-go));
      float cn = sf * c[q] + si * tanhf(gg);
      float hn = so * tanhf(cn);
      c[q] = cn;
      hS[m * HD + j] = (_Float16)hn;
      hout[((size_t)t * BB + m) * (2 * HD) + hoff + j] = (_Float16)hn;
    }
    __syncthreads();
  }
}

// ---------------------------------------------------------------------------
// FC: logits[S*B, 20] = h1[S*B, 1024] @ fc_w[20,1024]^T + fc_b  (tiny: 0.3GF)
// ---------------------------------------------------------------------------
__global__ void k_fc(const _Float16* __restrict__ h1, const float* __restrict__ fcw,
                     const float* __restrict__ fcb, float* __restrict__ logits) {
  int idx = blockIdx.x * blockDim.x + threadIdx.x;
  if (idx >= SS * BB * LL) return;
  int l = idx % LL;
  int row = idx / LL;
  const _Float16* hp = h1 + (size_t)row * (2 * HD);
  const float* wp = fcw + (size_t)l * (2 * HD);
  float acc = fcb[l];
  for (int k = 0; k < 2 * HD; ++k) acc += (float)hp[k] * wp[k];
  logits[idx] = acc;
}

// ---------------------------------------------------------------------------
// CRF NLL, one block (one wave32) per batch sample.
// ---------------------------------------------------------------------------
__global__ __launch_bounds__(32)
void k_crf(const float* __restrict__ logits, const int* __restrict__ labels,
           const int* __restrict__ mask, const float* __restrict__ trans,
           const float* __restrict__ sv, const float* __restrict__ ev,
           float* __restrict__ nll) {
  int b = blockIdx.x;
  int l = threadIdx.x;
  __shared__ float alpha[LL];
  if (l < LL) alpha[l] = logits[(size_t)b * LL + l] + sv[l];
  __syncthreads();

  for (int t = 1; t < SS; ++t) {
    float nv = 0.f;
    if (l < LL) {
      float emit = logits[((size_t)t * BB + b) * LL + l];
      float mx = -1e30f;
      for (int f = 0; f < LL; ++f)
        mx = fmaxf(mx, alpha[f] + trans[f * LL + l] + emit);
      float sum = 0.f;
      for (int f = 0; f < LL; ++f)
        sum += __expf(alpha[f] + trans[f * LL + l] + emit - mx);
      nv = mx + __logf(sum);
      if (mask[b * SS + t] <= 0) nv = alpha[l];
    }
    __syncthreads();
    if (l < LL) alpha[l] = nv;
    __syncthreads();
  }

  if (l == 0) {
    float mx = -1e30f;
    for (int f = 0; f < LL; ++f) mx = fmaxf(mx, alpha[f] + ev[f]);
    float sum = 0.f;
    for (int f = 0; f < LL; ++f) sum += __expf(alpha[f] + ev[f] - mx);
    float norm = mx + __logf(sum);

    float emitsum = 0.f, transsum = 0.f;
    int prev = labels[b * SS];
    int msum = 0;
    for (int t = 0; t < SS; ++t) {
      int tag = labels[b * SS + t];
      int mk = mask[b * SS + t] > 0;
      float mf = mk ? 1.f : 0.f;
      msum += mk;
      emitsum += logits[((size_t)t * BB + b) * LL + tag] * mf;
      if (t > 0) transsum += trans[prev * LL + tag] * mf;
      prev = tag;
    }
    int lastIdx = msum - 1;
    if (lastIdx < 0) lastIdx = 0;
    int lastTag = labels[b * SS + lastIdx];
    float gold = emitsum + transsum + sv[labels[b * SS]] + ev[lastTag];
    nll[b] = norm - gold;
  }
}

__global__ void k_mean(const float* __restrict__ nll, float* __restrict__ out) {
  if (threadIdx.x == 0 && blockIdx.x == 0) {
    float s = 0.f;
    for (int i = 0; i < BB; ++i) s += nll[i];
    out[0] = s / (float)BB;
  }
}

// ---------------------------------------------------------------------------
// Host-side orchestration
// ---------------------------------------------------------------------------
extern "C" void kernel_launch(void* const* d_in, const int* in_sizes, int n_in,
                              void* d_out, int out_size, void* d_ws, size_t ws_size,
                              hipStream_t stream) {
  (void)in_sizes; (void)n_in; (void)out_size; (void)ws_size;
  const float* inputs = (const float*)d_in[0];
  const int*   labels = (const int*)d_in[1];
  const int*   mask   = (const int*)d_in[2];
  // order: l0f, l0b, l1f, l1b
  const float* w_ih[4] = {(const float*)d_in[3],  (const float*)d_in[7],
                          (const float*)d_in[11], (const float*)d_in[15]};
  const float* w_hh[4] = {(const float*)d_in[4],  (const float*)d_in[8],
                          (const float*)d_in[12], (const float*)d_in[16]};
  const float* b_ih[4] = {(const float*)d_in[5],  (const float*)d_in[9],
                          (const float*)d_in[13], (const float*)d_in[17]};
  const float* b_hh[4] = {(const float*)d_in[6],  (const float*)d_in[10],
                          (const float*)d_in[14], (const float*)d_in[18]};
  const float* fcw = (const float*)d_in[19];
  const float* fcb = (const float*)d_in[20];
  const float* trn = (const float*)d_in[21];
  const float* sv  = (const float*)d_in[22];
  const float* ev  = (const float*)d_in[23];

  char* ws = (char*)d_ws;
  size_t off = 0;
  auto alloc = [&](size_t bytes) -> char* {
    off = (off + 255) & ~(size_t)255;
    char* p = ws + off;
    off += bytes;
    return p;
  };

  _Float16* x16 = (_Float16*)alloc((size_t)SS * BB * HD * 2);
  int wihK[4] = {HD, HD, 2 * HD, 2 * HD};
  _Float16* wih16[4];
  for (int i = 0; i < 4; ++i) wih16[i] = (_Float16*)alloc((size_t)G4 * wihK[i] * 2);
  _Float16* whh16[4];
  for (int i = 0; i < 4; ++i) whh16[i] = (_Float16*)alloc((size_t)G4 * HD * 2);
  float* bias[4];
  for (int i = 0; i < 4; ++i) bias[i] = (float*)alloc((size_t)G4 * 4);
  float* xw_f = (float*)alloc((size_t)SS * BB * G4 * 4);
  float* xw_b = (float*)alloc((size_t)SS * BB * G4 * 4);
  _Float16* h0 = (_Float16*)alloc((size_t)SS * BB * 2 * HD * 2);
  _Float16* h1 = (_Float16*)alloc((size_t)SS * BB * 2 * HD * 2);
  float* logits = (float*)alloc((size_t)SS * BB * LL * 4);
  float* nll = (float*)alloc(BB * 4);

  // --- conversions ---
  {
    int n = SS * BB * HD;
    k_x_transpose_f16<<<(n + 255) / 256, 256, 0, stream>>>(inputs, x16);
  }
  for (int i = 0; i < 4; ++i) {
    int n = G4 * wihK[i];
    k_cvt_f16<<<(n + 255) / 256, 256, 0, stream>>>(w_ih[i], wih16[i], n);
    int n2 = G4 * HD;
    k_cvt_f16<<<(n2 + 255) / 256, 256, 0, stream>>>(w_hh[i], whh16[i], n2);
    k_bias_sum<<<(G4 + 255) / 256, 256, 0, stream>>>(b_ih[i], b_hh[i], bias[i], G4);
  }

  const dim3 gemmGrid(2048), gemmBlk(256);          // 16384 waves = 512 Mtiles x 32 Ngroups
  const size_t smem = (size_t)BB * G4 * 4 + (size_t)BB * HD * 2;  // 288 KB

  // --- layer 0 ---
  k_gemm_xw<<<gemmGrid, gemmBlk, 0, stream>>>(x16, HD, wih16[0], bias[0], xw_f);
  k_gemm_xw<<<gemmGrid, gemmBlk, 0, stream>>>(x16, HD, wih16[1], bias[1], xw_b);
  k_lstm_recur<<<2, 1024, smem, stream>>>(xw_f, xw_b, whh16[0], whh16[1], h0);

  // --- layer 1 ---
  k_gemm_xw<<<gemmGrid, gemmBlk, 0, stream>>>(h0, 2 * HD, wih16[2], bias[2], xw_f);
  k_gemm_xw<<<gemmGrid, gemmBlk, 0, stream>>>(h0, 2 * HD, wih16[3], bias[3], xw_b);
  k_lstm_recur<<<2, 1024, smem, stream>>>(xw_f, xw_b, whh16[2], whh16[3], h1);

  // --- head + CRF ---
  {
    int n = SS * BB * LL;
    k_fc<<<(n + 255) / 256, 256, 0, stream>>>(h1, fcw, fcb, logits);
  }
  k_crf<<<BB, 32, 0, stream>>>(logits, labels, mask, trn, sv, ev, nll);
  k_mean<<<1, 32, 0, stream>>>(nll, (float*)d_out);
}
